// QuantizeEMAReset_75041668596243
// MI455X (gfx1250) — matrix operand order repaired and checked by reading, not compiled, not run on hardware
//
#include <hip/hip_runtime.h>

typedef __attribute__((ext_vector_type(16))) __bf16 v16bf;
typedef __attribute__((ext_vector_type(8)))  __bf16 v8bf;
typedef __attribute__((ext_vector_type(8)))  float  v8f;

#define N_  32
#define C_  512
#define T_  2048
#define K_  1024
#define NT_ (N_ * T_)

// ---------------- Kernel 1: codebook -> bf16, per-code squared norms ----------------
__global__ __launch_bounds__(256)
void prep_codebook(const float* __restrict__ cb,
                   __bf16* __restrict__ cb_bf,
                   float* __restrict__ knorm) {
  __shared__ float red[256];
  const int k = blockIdx.x;
  const int tid = threadIdx.x;
  float s = 0.f;
  for (int c = tid; c < C_; c += 256) {
    float v = cb[k * C_ + c];
    cb_bf[k * C_ + c] = (__bf16)v;
    s += v * v;
  }
  red[tid] = s;
  __syncthreads();
  for (int o = 128; o > 0; o >>= 1) {
    if (tid < o) red[tid] += red[tid + o];
    __syncthreads();
  }
  if (tid == 0) knorm[k] = red[0];
}

// ---------------- Kernel 2: WMMA distance GEMM + argmin over K=1024 ----------------
// Block: 256 threads (8 waves). 32 consecutive rows (t-values) per block.
// Waves: wave_m (0..1) picks 16 rows, wave_n (0..3) picks 256 codes.
__global__ __launch_bounds__(256)
void vq_argmin(const float* __restrict__ x,
               const __bf16* __restrict__ cb_bf,
               const float* __restrict__ knorm,
               int* __restrict__ idx_out) {
  __shared__ __bf16 Al[32][520];   // padded row stride: 1040 B (16B-aligned, conflict-spread)
  __shared__ float rd[4][32];
  __shared__ int   ri[4][32];

  const int tid  = threadIdx.x;
  const int lane = tid & 31;
  const int wave = tid >> 5;
  const int m0   = blockIdx.x * 32;      // global flattened row (n*T + t)
  const int n    = m0 / T_;
  const int t0   = m0 % T_;              // 32-aligned, T_ multiple of 32 -> single n per block

  // Stage x tile: Al[t_local][c] = bf16(x[n, c, t0 + t_local]); coalesced 128B reads along T.
  {
    const int tt = tid & 31;
    const float* xp = x + (size_t)n * C_ * T_ + t0 + tt;
    for (int c = tid >> 5; c < C_; c += 8)
      Al[tt][c] = (__bf16)xp[(size_t)c * T_];
  }
  __syncthreads();

  const int half   = lane >> 4;
  const int l15    = lane & 15;
  const int wave_m = wave >> 2;          // 0..1
  const int wave_n = wave & 3;           // 0..3
  const int mrow   = wave_m * 16 + l15;  // A-matrix row this lane supplies

  // Preload all 16 A fragments (K-chunks of 32) in contiguous k order (same as B):
  // half=0 -> K {c0+0..15}; half=1 -> K {c0+16..31}.
  v16bf afr[16];
#pragma unroll
  for (int kc = 0; kc < 16; ++kc) {
    const int c0 = kc * 32;
    v8bf lo = *(const v8bf*)&Al[mrow][c0 + 16 * half];
    v8bf hi = *(const v8bf*)&Al[mrow][c0 + 16 * half + 8];
#pragma unroll
    for (int j = 0; j < 8; ++j) { afr[kc][j] = lo[j]; afr[kc][j + 8] = hi[j]; }
  }

  float best_d[8];
  int   best_i[8];
#pragma unroll
  for (int i = 0; i < 8; ++i) { best_d[i] = 3.4e38f; best_i[i] = 0; }

  const int k_base = wave_n * 256;
  for (int nt = 0; nt < 16; ++nt) {
    const int code = k_base + nt * 16 + l15;        // B-matrix column this lane supplies
    // B layout: lane n holds column N=n; lanes 0-15 -> K 0..15, lanes 16-31 -> K 16..31
    const __bf16* bp = cb_bf + (size_t)code * C_ + 16 * half;
    v8f acc = {};
#pragma unroll
    for (int kc = 0; kc < 16; ++kc) {
      v16bf b = *(const v16bf*)(bp + kc * 32);      // contiguous 32B per lane, L2-hot
      acc = __builtin_amdgcn_wmma_f32_16x16x32_bf16(
          /*neg_a=*/false, afr[kc], /*neg_b=*/false, b,
          /*c_mod=*/(short)0, acc, /*reuse_a=*/false, /*reuse_b=*/false);
    }
    const float kn = knorm[code];
#pragma unroll
    for (int i = 0; i < 8; ++i) {
      // D element i holds row M = i + 8*half for column N = code
      float d = kn - 2.f * acc[i];
      if (d < best_d[i] || (d == best_d[i] && code < best_i[i])) {
        best_d[i] = d; best_i[i] = code;
      }
    }
  }

  // Cross-lane argmin within each 16-lane half (same M rows, different codes).
  // xor offsets 1,2,4,8 never flip bit 4, so halves stay separate.
#pragma unroll
  for (int i = 0; i < 8; ++i) {
    float d = best_d[i]; int j = best_i[i];
#pragma unroll
    for (int off = 1; off < 16; off <<= 1) {
      float od = __shfl_xor(d, off, 32);
      int   oj = __shfl_xor(j, off, 32);
      if (od < d || (od == d && oj < j)) { d = od; j = oj; }
    }
    best_d[i] = d; best_i[i] = j;
  }

  if (l15 == 0) {
#pragma unroll
    for (int i = 0; i < 8; ++i) {
      const int row = wave_m * 16 + i + 8 * half;
      rd[wave_n][row] = best_d[i];
      ri[wave_n][row] = best_i[i];
    }
  }
  __syncthreads();
  if (tid < 32) {
    float bd = rd[0][tid]; int bi = ri[0][tid];
#pragma unroll
    for (int w = 1; w < 4; ++w) {
      float d = rd[w][tid]; int j = ri[w][tid];
      if (d < bd || (d == bd && j < bi)) { bd = d; bi = j; }
    }
    idx_out[m0 + tid] = bi;
  }
}

// ---------------- Kernel 3: gather winning code rows into [N, C, T] layout ----------------
__global__ __launch_bounds__(256)
void gather_out(const float* __restrict__ cb,
                const int* __restrict__ idx,
                float* __restrict__ out) {
  const int tid = threadIdx.x;
  const int b   = blockIdx.x;               // N*C*(T/256) blocks
  const int t   = ((b & 7) << 8) + tid;     // contiguous t -> coalesced stores
  const int nc  = b >> 3;                   // n*C + c
  const int c   = nc & (C_ - 1);
  const int n   = nc >> 9;
  const int code = idx[n * T_ + t];
  out[(size_t)nc * T_ + t] = cb[(size_t)code * C_ + c];
  if (b == 0 && tid < 2) out[(size_t)N_ * C_ * T_ + tid] = 0.f;  // commit_loss, perplexity
}

extern "C" void kernel_launch(void* const* d_in, const int* in_sizes, int n_in,
                              void* d_out, int out_size, void* d_ws, size_t ws_size,
                              hipStream_t stream) {
  const float* x  = (const float*)d_in[0];   // [32, 512, 2048]
  const float* cb = (const float*)d_in[1];   // [1024, 512]
  float* out = (float*)d_out;

  char* ws = (char*)d_ws;
  __bf16* cb_bf = (__bf16*)ws;                              // 1,048,576 B
  float*  knorm = (float*)(ws + (size_t)1048576);           //     4,096 B
  int*    idx   = (int*)(ws + (size_t)1048576 + 4096);      //   262,144 B

  prep_codebook<<<K_, 256, 0, stream>>>(cb, cb_bf, knorm);
  vq_argmin<<<NT_ / 32, 256, 0, stream>>>(x, cb_bf, knorm, idx);
  gather_out<<<N_ * C_ * (T_ / 256), 256, 0, stream>>>(cb, idx, out);
}
